// GNNEncoder_37684043055341
// MI455X (gfx1250) — matrix operand order repaired and checked by reading, not compiled, not run on hardware
//
#include <hip/hip_runtime.h>
#include <math.h>

typedef __attribute__((ext_vector_type(16))) _Float16 v16h;
typedef __attribute__((ext_vector_type(8)))  _Float16 v8h;
typedef __attribute__((ext_vector_type(8)))  float    v8f;

// ---------------------------------------------------------------------------
// Monotone uint encoding of float for atomicMax-based segment max.
// ---------------------------------------------------------------------------
__device__ __forceinline__ unsigned f32key(float f) {
  unsigned u = __float_as_uint(f);
  return (u & 0x80000000u) ? ~u : (u | 0x80000000u);
}
__device__ __forceinline__ float keyf32(unsigned k) {
  unsigned u = (k & 0x80000000u) ? (k ^ 0x80000000u) : ~k;
  return __uint_as_float(u);
}

// ---------------------------------------------------------------------------
// Repack f32 weights W[K,N] into f16 WMMA B-fragment layout:
//   Bfrag[ks][nt][lane][i]  (512 halves per 32x16 k-step/col-tile fragment)
//   lane<16: col = nt*16+lane, K = ks*32 + i      (i = 0..15)
//   lane>=16: col = nt*16+lane-16, K = ks*32+16+i
// Each GEMM lane then reads its 16 halves as one contiguous 32B chunk.
// ---------------------------------------------------------------------------
__global__ void repack_weights(const float* __restrict__ W,
                               _Float16* __restrict__ Bfrag,
                               int K, int N) {
  int tid = blockIdx.x * blockDim.x + threadIdx.x;
  int total = (K >> 5) * (N >> 4) * 512;
  if (tid >= total) return;
  int i    = tid & 15;
  int lane = (tid >> 4) & 31;
  int rest = tid >> 9;
  int ntiles = N >> 4;
  int nt = rest % ntiles;
  int ks = rest / ntiles;
  int col = nt * 16 + (lane & 15);
  int k   = ks * 32 + (lane < 16 ? 0 : 16) + i;
  Bfrag[tid] = (_Float16)W[(size_t)k * N + col];
}

// ---------------------------------------------------------------------------
// Activation converts (f32 -> f16), optionally fused with ReLU.
// ---------------------------------------------------------------------------
__global__ void cvt_f16(const float* __restrict__ in, _Float16* __restrict__ out,
                        long long n) {
  long long tid = (long long)blockIdx.x * blockDim.x + threadIdx.x;
  if (tid >= n) return;
  out[tid] = (_Float16)in[tid];
}
__global__ void cvt_relu_f16(const float* __restrict__ in, _Float16* __restrict__ out,
                             long long n) {
  long long tid = (long long)blockIdx.x * blockDim.x + threadIdx.x;
  if (tid >= n) return;
  float v = in[tid];
  out[tid] = (_Float16)(v > 0.0f ? v : 0.0f);
}

// ---------------------------------------------------------------------------
// WMMA GEMM: C[M,N] = Ah[M,K](f16) * W(f16 prepacked fragments) + bias[N]
// One wave -> 32x32 output block: 2 A fragments x 2 B fragments = 4 WMMAs/k-step.
// Requires M % 32 == 0, K % 32 == 0, N % 32 == 0 (true for all layers).
// ---------------------------------------------------------------------------
__global__ void gemm_wmma_bias(const _Float16* __restrict__ Ah,
                               const _Float16* __restrict__ Bfrag,
                               const float* __restrict__ bias,
                               float* __restrict__ C,
                               int M, int K, int N) {
  const int wavesPerBlock = blockDim.x >> 5;
  const int wave = blockIdx.x * wavesPerBlock + ((int)threadIdx.x >> 5);
  const int lane = (int)threadIdx.x & 31;
  const int stripsN = N >> 5;           // 32-wide column strips
  const int blocksM = M >> 5;           // 32-row blocks
  const int tM = wave / stripsN;
  const int tN = wave - tM * stripsN;
  if (tM >= blocksM) return;            // whole-wave uniform exit: EXEC stays full

  const bool hi   = lane >= 16;
  const int  row0 = tM * 32 + (lane & 15);
  const int  row1 = row0 + 16;
  const int  ntiles = N >> 4;
  const int  nt0 = tN * 2;

  // A-fragment half offsets within a 32-wide k-step (matches B repack):
  // lane<16 holds K chunks {0..7, 8..15}; lane>=16 holds {16..23, 24..31}.
  const int ka0 = hi ? 16 : 0;
  const int ka1 = hi ? 24 : 8;

  v8f acc00 = {}, acc01 = {}, acc10 = {}, acc11 = {};

  for (int k0 = 0; k0 < K; k0 += 32) {
    const _Float16* ar0 = Ah + (size_t)row0 * K + k0;
    const _Float16* ar1 = Ah + (size_t)row1 * K + k0;
    v8h a0lo = *(const v8h*)(ar0 + ka0);
    v8h a0hi = *(const v8h*)(ar0 + ka1);
    v8h a1lo = *(const v8h*)(ar1 + ka0);
    v8h a1hi = *(const v8h*)(ar1 + ka1);
    v16h a0 = __builtin_shufflevector(a0lo, a0hi, 0,1,2,3,4,5,6,7,8,9,10,11,12,13,14,15);
    v16h a1 = __builtin_shufflevector(a1lo, a1hi, 0,1,2,3,4,5,6,7,8,9,10,11,12,13,14,15);

    const _Float16* bf = Bfrag + ((size_t)(k0 >> 5) * ntiles + nt0) * 512 + lane * 16;
    v16h b0 = *(const v16h*)bf;           // contiguous 32B -> 2x b128
    v16h b1 = *(const v16h*)(bf + 512);

    acc00 = __builtin_amdgcn_wmma_f32_16x16x32_f16(false, a0, false, b0, (short)0, acc00, false, false);
    acc01 = __builtin_amdgcn_wmma_f32_16x16x32_f16(false, a0, false, b1, (short)0, acc01, false, false);
    acc10 = __builtin_amdgcn_wmma_f32_16x16x32_f16(false, a1, false, b0, (short)0, acc10, false, false);
    acc11 = __builtin_amdgcn_wmma_f32_16x16x32_f16(false, a1, false, b1, (short)0, acc11, false, false);
  }

  // C/D layout: VGPR r -> row (+8 for hi half-lanes), lane&15 -> column.
  const int col0 = tN * 32 + (lane & 15);
  const int col1 = col0 + 16;
  const float bv0 = bias[col0];
  const float bv1 = bias[col1];
  const int rb0 = tM * 32 + (hi ? 8 : 0);
  const int rb1 = rb0 + 16;
#pragma unroll
  for (int r = 0; r < 8; ++r) {
    C[(size_t)(rb0 + r) * N + col0] = acc00[r] + bv0;
    C[(size_t)(rb0 + r) * N + col1] = acc01[r] + bv1;
    C[(size_t)(rb1 + r) * N + col0] = acc10[r] + bv0;
    C[(size_t)(rb1 + r) * N + col1] = acc11[r] + bv1;
  }
}

// ---------------------------------------------------------------------------
// Per-node softmax stats init: max-key = 0 (== -inf key), denom = 0.
// ---------------------------------------------------------------------------
__global__ void init_stats(unsigned* __restrict__ amaxKey,
                           float* __restrict__ denom, int n) {
  int tid = blockIdx.x * blockDim.x + threadIdx.x;
  if (tid >= n) return;
  amaxKey[tid] = 0u;
  denom[tid]   = 0.0f;
}

// ---------------------------------------------------------------------------
// Pass 1: alpha[e,h] = <q[dst,h,:], k[src,h,:]> / sqrt(C); segment max.
// ---------------------------------------------------------------------------
__global__ void edge_alpha(const float* __restrict__ Q,
                           const float* __restrict__ Kf,
                           const int* __restrict__ src,
                           const int* __restrict__ dst,
                           float* __restrict__ alpha,
                           unsigned* __restrict__ amaxKey,
                           int E, int H, int C) {
  int tid = blockIdx.x * blockDim.x + threadIdx.x;
  if (tid >= E * H) return;
  int e = tid / H;
  int h = tid - e * H;
  int s = src[e];
  int d = dst[e];
  int dout = H * C;
  const float* qp = Q  + (size_t)d * dout + (size_t)h * C;
  const float* kp = Kf + (size_t)s * dout + (size_t)h * C;
  float acc = 0.0f;
  for (int c = 0; c < C; ++c) acc += qp[c] * kp[c];
  acc *= rsqrtf((float)C);
  alpha[tid] = acc;
  atomicMax(&amaxKey[(size_t)d * H + h], f32key(acc));
}

// ---------------------------------------------------------------------------
// Pass 2: ex = exp(alpha - max[dst]); alpha <- ex; denom[dst] += ex.
// ---------------------------------------------------------------------------
__global__ void edge_exp(float* __restrict__ alpha,
                         const unsigned* __restrict__ amaxKey,
                         float* __restrict__ denom,
                         const int* __restrict__ dst,
                         int E, int H) {
  int tid = blockIdx.x * blockDim.x + threadIdx.x;
  if (tid >= E * H) return;
  int e = tid / H;
  int h = tid - e * H;
  int d = dst[e];
  float m  = keyf32(amaxKey[(size_t)d * H + h]);
  float ex = __expf(alpha[tid] - m);
  alpha[tid] = ex;
  atomicAdd(&denom[(size_t)d * H + h], ex);
}

// ---------------------------------------------------------------------------
// Pass 3: agg[dst, i] += v[src, i] * ex / (denom[dst,h] + 1e-16).
// ---------------------------------------------------------------------------
__global__ void edge_aggregate(const float* __restrict__ Vf,
                               const float* __restrict__ alpha,
                               const float* __restrict__ denom,
                               const int* __restrict__ src,
                               const int* __restrict__ dst,
                               float* __restrict__ agg,
                               int E, int H, int C) {
  const int dout = H * C;
  long long tid = (long long)blockIdx.x * blockDim.x + threadIdx.x;
  long long total = (long long)E * dout;
  if (tid >= total) return;
  int e = (int)(tid / dout);
  int i = (int)(tid - (long long)e * dout);
  int h = i / C;
  int s = src[e];
  int d = dst[e];
  float attn = alpha[(size_t)e * H + h] / (denom[(size_t)d * H + h] + 1e-16f);
  atomicAdd(&agg[(size_t)d * dout + i], Vf[(size_t)s * dout + i] * attn);
}

// ---------------------------------------------------------------------------
// Host launcher.
// ---------------------------------------------------------------------------
extern "C" void kernel_launch(void* const* d_in, const int* in_sizes, int n_in,
                              void* d_out, int out_size, void* d_ws, size_t ws_size,
                              hipStream_t stream) {
  (void)n_in; (void)out_size; (void)ws_size;

  const float* x  = (const float*)d_in[0];
  const int*   ei = (const int*)d_in[1];
  const int Nn = in_sizes[0] / 64;    // N_NODES
  const int E  = in_sizes[1] / 2;     // N_EDGES
  const int* src = ei;
  const int* dst = ei + E;

  // Params: for each layer l: Wq,bq,Wk,bk,Wv,bv,Ws,bs
  const float* W[3][4];
  const float* Bp[3][4];
  int idx = 2;
  for (int l = 0; l < 3; ++l)
    for (int m = 0; m < 4; ++m) {
      W[l][m]  = (const float*)d_in[idx++];
      Bp[l][m] = (const float*)d_in[idx++];
    }

  const int Hs[3]   = {7, 4, 1};
  const int Cs[3]   = {32, 32, 64};
  const int dins[3] = {64, 224, 128};

  // Workspace layout (float units; every size below is a multiple of 16 floats,
  // so all sub-buffers stay 64B-aligned for b128 vector loads).
  float* ws = (float*)d_ws;
  size_t o = 0;
  float*    Qb    = ws + o; o += (size_t)Nn * 224;
  float*    Kb    = ws + o; o += (size_t)Nn * 224;
  float*    Vb    = ws + o; o += (size_t)Nn * 224;
  float*    aggA  = ws + o; o += (size_t)Nn * 224;   // layer-1 out / layer-2 in (f32)
  float*    aggB  = ws + o; o += (size_t)Nn * 128;   // layer-2 out / layer-3 in (f32)
  float*    alpha = ws + o; o += (size_t)E * 7;
  unsigned* amax  = (unsigned*)(ws + o); o += (size_t)Nn * 7;
  float*    denom = ws + o; o += (size_t)Nn * 7;
  _Float16* Ah    = (_Float16*)(ws + o); o += (size_t)Nn * 112;      // N x 224 halves
  const size_t fragSlot = (size_t)7 * 14 * 512;                      // max (K/32)*(N/16)*512
  _Float16* Bfr   = (_Float16*)(ws + o); o += (4 * fragSlot + 1) / 2;

  const float* layerInF32[3] = { x, aggA, aggB };
  float*       layerOut[3]   = { aggA, aggB, (float*)d_out };

  for (int l = 0; l < 3; ++l) {
    const int H = Hs[l], C = Cs[l], din = dins[l];
    const int dout = H * C;

    // --- convert this layer's input activations to f16 (ReLU fused for l>0) ---
    const long long nIn = (long long)Nn * din;
    if (l == 0)
      cvt_f16<<<(int)((nIn + 255) / 256), 256, 0, stream>>>(layerInF32[l], Ah, nIn);
    else
      cvt_relu_f16<<<(int)((nIn + 255) / 256), 256, 0, stream>>>(layerInF32[l], Ah, nIn);

    // --- repack the 4 weight matrices into WMMA fragment layout ---
    const int fragHalves = (din >> 5) * (dout >> 4) * 512;
    for (int m = 0; m < 4; ++m)
      repack_weights<<<(fragHalves + 255) / 256, 256, 0, stream>>>(
          W[l][m], Bfr + (size_t)m * fragSlot, din, dout);

    // --- dense projections (WMMA) ---
    const int totalWaves = (Nn >> 5) * (dout >> 5);
    const int wavesPerBlock = 4;                  // 128 threads = 4 wave32
    const int gBlocks = (totalWaves + wavesPerBlock - 1) / wavesPerBlock;
    gemm_wmma_bias<<<gBlocks, 128, 0, stream>>>(Ah, Bfr + 0 * fragSlot, Bp[l][0], Qb,          Nn, din, dout);
    gemm_wmma_bias<<<gBlocks, 128, 0, stream>>>(Ah, Bfr + 1 * fragSlot, Bp[l][1], Kb,          Nn, din, dout);
    gemm_wmma_bias<<<gBlocks, 128, 0, stream>>>(Ah, Bfr + 2 * fragSlot, Bp[l][2], Vb,          Nn, din, dout);
    // Skip connection written straight into the aggregation target.
    gemm_wmma_bias<<<gBlocks, 128, 0, stream>>>(Ah, Bfr + 3 * fragSlot, Bp[l][3], layerOut[l], Nn, din, dout);

    // --- segment softmax + message aggregation ---
    const int nh = Nn * H;
    init_stats<<<(nh + 255) / 256, 256, 0, stream>>>(amax, denom, nh);

    const int eh = E * H;
    edge_alpha<<<(eh + 255) / 256, 256, 0, stream>>>(Qb, Kb, src, dst, alpha, amax, E, H, C);
    edge_exp<<<(eh + 255) / 256, 256, 0, stream>>>(alpha, amax, denom, dst, E, H);

    const long long total = (long long)E * dout;
    edge_aggregate<<<(int)((total + 255) / 256), 256, 0, stream>>>(
        Vb, alpha, denom, src, dst, layerOut[l], E, H, C);
    // No separate ReLU pass: fused into next layer's cvt_relu_f16.
  }
}
